// SelfAttentionLayer_47321949667362
// MI455X (gfx1250) — compile-verified, matrix operand order checked by reading, not run on hardware
//
#include <hip/hip_runtime.h>
#include <hip/hip_bf16.h>

// ---------------- problem constants ----------------
#define BB 8
#define TT 2048
#define DD 1024
#define MM (BB * TT)        // 16384 rows
#define N3 (3 * DD)         // 3072
#define SCALE 0.022097086912079608f   // 1/sqrt(T)
#define NEG_BIG (-3.0e38f)

typedef __attribute__((ext_vector_type(16))) __bf16 v16bf;
typedef __attribute__((ext_vector_type(8)))  __bf16 v8bf;
typedef __attribute__((ext_vector_type(8)))  float  v8f;

#define WMMA_BF16(A, Bm, C) \
  __builtin_amdgcn_wmma_f32_16x16x32_bf16(false, (A), false, (Bm), (short)0, (C), false, false)

// round-to-nearest-even f32 -> bf16
__device__ __forceinline__ __bf16 f32_to_bf16(float f) {
  unsigned u = __builtin_bit_cast(unsigned, f);
  unsigned r = u + 0x7FFFu + ((u >> 16) & 1u);
  unsigned short h = (unsigned short)(r >> 16);
  return __builtin_bit_cast(__bf16, h);
}

// Load one 16x32 bf16 operand tile (A-layout; B uses the same striping with
// "row" = B column). p0 -> element (row0, k0); ld = row stride in elements.
// lane<16 : row=lane,    K = {0..7, 16..23}
// lane>=16: row=lane-16, K = {8..15, 24..31}
__device__ __forceinline__ v16bf load_tile16x32(const __bf16* p0, int ld, int lane) {
  int r = lane & 15;
  int h = (lane >> 4) & 1;
  const __bf16* p = p0 + (size_t)r * ld + h * 8;
  v8bf lo = *(const v8bf*)(p);
  v8bf hi = *(const v8bf*)(p + 16);
  v16bf out;
#pragma unroll
  for (int i = 0; i < 8; ++i) { out[i] = lo[i]; out[8 + i] = hi[i]; }
  return out;
}

// ---------------- conversion / transpose prep ----------------
__global__ void k_cvt_f32_bf16(const float* __restrict__ s, __bf16* __restrict__ d, int n) {
  int i = blockIdx.x * blockDim.x + threadIdx.x;
  int stride = gridDim.x * blockDim.x;
  for (; i < n; i += stride) d[i] = f32_to_bf16(s[i]);
}

// s: [R][C] f32  ->  d: [C][R] bf16 (transposed)
__global__ void k_cvtT_f32_bf16(const float* __restrict__ s, __bf16* __restrict__ d, int R, int C) {
  int n = R * C;
  int i = blockIdx.x * blockDim.x + threadIdx.x;
  int stride = gridDim.x * blockDim.x;
  for (; i < n; i += stride) {
    int r = i / C, c = i - r * C;
    d[(size_t)c * R + r] = f32_to_bf16(s[i]);
  }
}

// ---- shared 32x64 double-buffered mainloop over K = DD ----
// Ab -> (m0, 0) with row stride DD ; Bb -> (n0, 0) with row stride DD.
// acc[mi*4 + j] accumulates tile (m0+16*mi, n0+16*j).
__device__ __forceinline__ void mainloop_32x64(const __bf16* __restrict__ Ab,
                                               const __bf16* __restrict__ Bb,
                                               int lane, v8f acc[8]) {
  v16bf a0 = load_tile16x32(Ab, DD, lane);
  v16bf a1 = load_tile16x32(Ab + (size_t)16 * DD, DD, lane);
  v16bf b0 = load_tile16x32(Bb, DD, lane);
  v16bf b1 = load_tile16x32(Bb + (size_t)16 * DD, DD, lane);
  v16bf b2 = load_tile16x32(Bb + (size_t)32 * DD, DD, lane);
  v16bf b3 = load_tile16x32(Bb + (size_t)48 * DD, DD, lane);
  for (int k = 32; k < DD; k += 32) {
    v16bf na0 = load_tile16x32(Ab + k, DD, lane);
    v16bf na1 = load_tile16x32(Ab + (size_t)16 * DD + k, DD, lane);
    v16bf nb0 = load_tile16x32(Bb + k, DD, lane);
    v16bf nb1 = load_tile16x32(Bb + (size_t)16 * DD + k, DD, lane);
    v16bf nb2 = load_tile16x32(Bb + (size_t)32 * DD + k, DD, lane);
    v16bf nb3 = load_tile16x32(Bb + (size_t)48 * DD + k, DD, lane);
    acc[0] = WMMA_BF16(a0, b0, acc[0]);
    acc[1] = WMMA_BF16(a0, b1, acc[1]);
    acc[2] = WMMA_BF16(a0, b2, acc[2]);
    acc[3] = WMMA_BF16(a0, b3, acc[3]);
    acc[4] = WMMA_BF16(a1, b0, acc[4]);
    acc[5] = WMMA_BF16(a1, b1, acc[5]);
    acc[6] = WMMA_BF16(a1, b2, acc[6]);
    acc[7] = WMMA_BF16(a1, b3, acc[7]);
    a0 = na0; a1 = na1; b0 = nb0; b1 = nb1; b2 = nb2; b3 = nb3;
  }
  acc[0] = WMMA_BF16(a0, b0, acc[0]);
  acc[1] = WMMA_BF16(a0, b1, acc[1]);
  acc[2] = WMMA_BF16(a0, b2, acc[2]);
  acc[3] = WMMA_BF16(a0, b3, acc[3]);
  acc[4] = WMMA_BF16(a1, b0, acc[4]);
  acc[5] = WMMA_BF16(a1, b1, acc[5]);
  acc[6] = WMMA_BF16(a1, b2, acc[6]);
  acc[7] = WMMA_BF16(a1, b3, acc[7]);
}

// ---------------- QKV GEMM:  [M,K]x[K,N3]+bias -> Kb,Qb (row major), Vt (transposed) ----
__global__ void __launch_bounds__(256, 1)
k_qkv_gemm(const __bf16* __restrict__ Xb,   // [MM][DD]
           const __bf16* __restrict__ WT,   // [N3][DD]  (W_kqv transposed)
           const float*  __restrict__ bias, // [N3]
           __bf16* __restrict__ Kb,         // [BB][TT][DD]
           __bf16* __restrict__ Qb,         // [BB][TT][DD]
           __bf16* __restrict__ Vt)         // [BB][DD][TT]
{
  const int nWavesN = N3 / 64;                       // 48
  int wave = blockIdx.x * (blockDim.x >> 5) + (threadIdx.x >> 5);
  int lane = threadIdx.x & 31;
  int mt = wave / nWavesN;                           // 32-row tiles
  int nt = wave - mt * nWavesN;
  if (mt >= MM / 32) return;                         // wave-uniform
  int m0 = mt * 32, n0 = nt * 64;

  v8f acc[8] = {v8f{}, v8f{}, v8f{}, v8f{}, v8f{}, v8f{}, v8f{}, v8f{}};
  mainloop_32x64(Xb + (size_t)m0 * DD, WT + (size_t)n0 * DD, lane, acc);

  int nlo = lane & 15, mhi = (lane >> 4) << 3;
#pragma unroll
  for (int mi = 0; mi < 2; ++mi) {
#pragma unroll
    for (int j = 0; j < 4; ++j) {
      int nj  = n0 + 16 * j;
      int col = nj + nlo;
      float bv = bias[col];
#pragma unroll
      for (int i = 0; i < 8; ++i) {
        int m = m0 + 16 * mi + mhi + i;
        int b = m >> 11;            // /TT
        int t = m & (TT - 1);
        __bf16 h = f32_to_bf16(acc[mi * 4 + j][i] + bv);
        if (nj < DD)            Kb[((size_t)(b * TT + t)) * DD + col] = h;
        else if (nj < 2 * DD)   Qb[((size_t)(b * TT + t)) * DD + (col - DD)] = h;
        else                    Vt[((size_t)(b * DD + (col - 2 * DD))) * TT + t] = h;
      }
    }
  }
}

// ---------------- fused causal attention ----------------
// one workgroup (256 thr = 8 waves) per (batch, 16-row query tile)
#define SROW (TT + 8)     // f32 score row stride (pad vs bank conflicts)
#define PROW (TT + 16)    // bf16 prob row stride (16B aligned rows)
#define SM_S_BYTES (16 * SROW * 4)
#define SM_P_BYTES (16 * PROW * 2)
#define SM_RED_BYTES (16 * 16 * 4)
#define SM_TOTAL (SM_S_BYTES + SM_P_BYTES + SM_RED_BYTES + 2 * 16 * 4)

__global__ void __launch_bounds__(256, 1)
k_attn(const __bf16* __restrict__ Qb,  // [BB][TT][DD]
       const __bf16* __restrict__ Kb,  // [BB][TT][DD]
       const __bf16* __restrict__ Vt,  // [BB][DD][TT]
       __bf16* __restrict__ Ob)        // [BB][TT][DD]
{
  extern __shared__ char smem[];
  float*  S      = (float*)smem;                                   // [16][SROW]
  __bf16* P      = (__bf16*)(smem + SM_S_BYTES);                   // [16][PROW]
  float*  red    = (float*)(smem + SM_S_BYTES + SM_P_BYTES);       // [16][16]
  float*  rowmax = (float*)(smem + SM_S_BYTES + SM_P_BYTES + SM_RED_BYTES);
  float*  rowsum = rowmax + 16;

  int b  = blockIdx.x >> 7;            // / (TT/16)
  int qt = blockIdx.x & 127;
  int q0 = qt * 16;
  int wid = threadIdx.x >> 5, lane = threadIdx.x & 31;
  int nlo = lane & 15, mhi = (lane >> 4) << 3;

  int nst    = qt + 1;                 // causal key tiles needed
  int nstPad = (nst + 1) & ~1;         // pad to even (32-wide chunks)
  int nch    = nstPad >> 1;
  int ncol   = nstPad * 16;

  const __bf16* Qbase = Qb + ((size_t)(b * TT + q0)) * DD;
  const __bf16* Kbase = Kb + ((size_t)b * TT) * DD;
  const __bf16* Vbase = Vt + ((size_t)b * DD) * TT;

  // ---- phase 1: S = scale * Q K^T with causal mask, into LDS ----
  // each wave handles 32-key chunks (2 stiles) sharing the Q (A) loads
  for (int ch = wid; ch < nch; ch += 8) {             // wave-uniform
    int s0 = ch * 32;
    if (s0 > q0 + 15) {                               // fully masked chunk
#pragma unroll
      for (int i = 0; i < 8; ++i) {
        S[(mhi + i) * SROW + s0 + nlo]      = NEG_BIG;
        S[(mhi + i) * SROW + s0 + 16 + nlo] = NEG_BIG;
      }
      continue;
    }
    v8f c0 = {}, c1 = {};
    v16bf a  = load_tile16x32(Qbase, DD, lane);
    v16bf b0 = load_tile16x32(Kbase + (size_t)s0 * DD, DD, lane);
    v16bf b1 = load_tile16x32(Kbase + (size_t)(s0 + 16) * DD, DD, lane);
    for (int k = 32; k < DD; k += 32) {
      v16bf na  = load_tile16x32(Qbase + k, DD, lane);
      v16bf nb0 = load_tile16x32(Kbase + (size_t)s0 * DD + k, DD, lane);
      v16bf nb1 = load_tile16x32(Kbase + (size_t)(s0 + 16) * DD + k, DD, lane);
      c0 = WMMA_BF16(a, b0, c0);
      c1 = WMMA_BF16(a, b1, c1);
      a = na; b0 = nb0; b1 = nb1;
    }
    c0 = WMMA_BF16(a, b0, c0);
    c1 = WMMA_BF16(a, b1, c1);
#pragma unroll
    for (int i = 0; i < 8; ++i) {
      int m = mhi + i;
      int col0 = s0 + nlo, col1 = s0 + 16 + nlo;
      float v0 = c0[i] * SCALE;
      float v1 = c1[i] * SCALE;
      if (col0 > q0 + m) v0 = NEG_BIG;
      if (col1 > q0 + m) v1 = NEG_BIG;
      S[m * SROW + col0] = v0;
      S[m * SROW + col1] = v1;
    }
  }
  __syncthreads();

  // ---- phase 2: softmax rows in LDS, write normalized P (bf16) ----
  {
    int r  = threadIdx.x >> 4;     // 0..15
    int cg = threadIdx.x & 15;
    float mx = NEG_BIG;
    for (int c = cg; c < ncol; c += 16) mx = fmaxf(mx, S[r * SROW + c]);
    red[r * 16 + cg] = mx;
    __syncthreads();
    if (cg == 0) {
      float m = red[r * 16];
#pragma unroll
      for (int j = 1; j < 16; ++j) m = fmaxf(m, red[r * 16 + j]);
      rowmax[r] = m;
    }
    __syncthreads();
    float rm = rowmax[r];
    float sacc = 0.f;
    for (int c = cg; c < ncol; c += 16) sacc += __expf(S[r * SROW + c] - rm);
    red[r * 16 + cg] = sacc;
    __syncthreads();
    if (cg == 0) {
      float s = red[r * 16];
#pragma unroll
      for (int j = 1; j < 16; ++j) s += red[r * 16 + j];
      rowsum[r] = s;
    }
    __syncthreads();
    float inv = 1.f / rowsum[r];
    for (int c = cg; c < ncol; c += 16)
      P[r * PROW + c] = f32_to_bf16(__expf(S[r * SROW + c] - rm) * inv);
  }
  __syncthreads();

  // ---- phase 3: O = P V ; each wave owns a 128-wide d slice ----
  int d0 = wid * 128;
  v8f o[8] = {v8f{}, v8f{}, v8f{}, v8f{}, v8f{}, v8f{}, v8f{}, v8f{}};
  for (int ch = 0; ch < nch; ++ch) {
    int s0 = ch * 32;
    v16bf a = load_tile16x32((const __bf16*)P + s0, PROW, lane);   // ds_load
    // two groups of 4 V tiles: clause loads, bounded register pressure
    v16bf vb0[4], vb1[4];
#pragma unroll
    for (int j = 0; j < 4; ++j)
      vb0[j] = load_tile16x32(Vbase + (size_t)(d0 + 16 * j) * TT + s0, TT, lane);
#pragma unroll
    for (int j = 0; j < 4; ++j) o[j] = WMMA_BF16(a, vb0[j], o[j]);
#pragma unroll
    for (int j = 0; j < 4; ++j)
      vb1[j] = load_tile16x32(Vbase + (size_t)(d0 + 64 + 16 * j) * TT + s0, TT, lane);
#pragma unroll
    for (int j = 0; j < 4; ++j) o[4 + j] = WMMA_BF16(a, vb1[j], o[4 + j]);
  }
#pragma unroll
  for (int j = 0; j < 8; ++j) {
    int d = d0 + 16 * j + nlo;
#pragma unroll
    for (int i = 0; i < 8; ++i) {
      int m = mhi + i;
      Ob[((size_t)(b * TT + q0 + m)) * DD + d] = f32_to_bf16(o[j][i]);
    }
  }
}

// ---------------- output projection GEMM: f32 out + bias ----------------
__global__ void __launch_bounds__(256, 1)
k_proj_gemm(const __bf16* __restrict__ Ab,   // [MM][DD]
            const __bf16* __restrict__ WT,   // [DD][DD] transposed
            const float*  __restrict__ bias, // [DD]
            float* __restrict__ out)         // [MM][DD]
{
  const int nWavesN = DD / 64;                        // 16
  int wave = blockIdx.x * (blockDim.x >> 5) + (threadIdx.x >> 5);
  int lane = threadIdx.x & 31;
  int mt = wave / nWavesN;
  int nt = wave - mt * nWavesN;
  if (mt >= MM / 32) return;
  int m0 = mt * 32, n0 = nt * 64;

  v8f acc[8] = {v8f{}, v8f{}, v8f{}, v8f{}, v8f{}, v8f{}, v8f{}, v8f{}};
  mainloop_32x64(Ab + (size_t)m0 * DD, WT + (size_t)n0 * DD, lane, acc);

  int nlo = lane & 15, mhi = (lane >> 4) << 3;
#pragma unroll
  for (int mi = 0; mi < 2; ++mi) {
#pragma unroll
    for (int j = 0; j < 4; ++j) {
      int col = n0 + 16 * j + nlo;
      float bv = bias[col];
#pragma unroll
      for (int i = 0; i < 8; ++i) {
        int m = m0 + 16 * mi + mhi + i;
        out[(size_t)m * DD + col] = acc[mi * 4 + j][i] + bv;
      }
    }
  }
}

// ---------------- host launcher ----------------
extern "C" void kernel_launch(void* const* d_in, const int* in_sizes, int n_in,
                              void* d_out, int out_size, void* d_ws, size_t ws_size,
                              hipStream_t stream) {
  const float* x      = (const float*)d_in[0];
  const float* W_kqv  = (const float*)d_in[1];
  const float* b_kqv  = (const float*)d_in[2];
  const float* W_proj = (const float*)d_in[3];
  const float* b_proj = (const float*)d_in[4];
  float* out = (float*)d_out;

  // workspace carve-up (bf16 buffers)
  char* p = (char*)d_ws;
  __bf16* Xb     = (__bf16*)p; p += (size_t)MM * DD * 2;   // 32 MB
  __bf16* WkqvT  = (__bf16*)p; p += (size_t)N3 * DD * 2;   // 6 MB
  __bf16* WprojT = (__bf16*)p; p += (size_t)DD * DD * 2;   // 2 MB
  __bf16* Kb     = (__bf16*)p; p += (size_t)MM * DD * 2;   // 32 MB
  __bf16* Qb     = (__bf16*)p; p += (size_t)MM * DD * 2;   // 32 MB
  __bf16* Vt     = (__bf16*)p; p += (size_t)MM * DD * 2;   // 32 MB  [B][D][T]
  __bf16* Ob     = (__bf16*)p; p += (size_t)MM * DD * 2;   // 32 MB
  (void)ws_size; (void)in_sizes; (void)n_in; (void)out_size;

  // prep: conversions / transposes
  k_cvt_f32_bf16 <<<4096, 256, 0, stream>>>(x, Xb, MM * DD);
  k_cvtT_f32_bf16<<<4096, 256, 0, stream>>>(W_kqv,  WkqvT,  DD, N3);
  k_cvtT_f32_bf16<<<2048, 256, 0, stream>>>(W_proj, WprojT, DD, DD);

  // QKV projection: 32x64 tile per wave
  {
    int waves = (MM / 32) * (N3 / 64);               // 24576
    k_qkv_gemm<<<waves / 8, 256, 0, stream>>>(Xb, WkqvT, b_kqv, Kb, Qb, Vt);
  }

  // fused causal attention (dynamic LDS ~194 KB of the 320 KB WGP pool)
  k_attn<<<BB * (TT / 16), 256, SM_TOTAL, stream>>>(Qb, Kb, Vt, Ob);

  // output projection: 32x64 tile per wave
  {
    int waves = (MM / 32) * (DD / 64);               // 8192
    k_proj_gemm<<<waves / 8, 256, 0, stream>>>(Ob, WprojT, b_proj, out);
  }
}
